// TopKActivationMLP_27719718928757
// MI455X (gfx1250) — matrix operand order repaired and checked by reading, not compiled, not run on hardware
//
#include <hip/hip_runtime.h>
#include <hip/hip_bf16.h>

typedef __bf16 bf16_t;
typedef __attribute__((ext_vector_type(16))) __bf16 v16bf;
typedef __attribute__((ext_vector_type(8)))  __bf16 v8bf;
typedef __attribute__((ext_vector_type(8)))  float  v8f;

// ---------------------------------------------------------------------------
// float -> bf16 (round-to-nearest-even) as raw bits
// ---------------------------------------------------------------------------
__device__ __forceinline__ unsigned short f2bf_bits(float f) {
    unsigned u = __float_as_uint(f);
    unsigned r = u + 0x7FFFu + ((u >> 16) & 1u);
    return (unsigned short)(r >> 16);
}

// ---------------------------------------------------------------------------
// Elementwise f32 -> bf16 with zero padding (dst length ndst >= nsrc)
// ---------------------------------------------------------------------------
__global__ void cvt_f32_to_bf16_pad(const float* __restrict__ src,
                                    unsigned short* __restrict__ dst,
                                    long nsrc, long ndst) {
    long i = (long)blockIdx.x * blockDim.x + threadIdx.x;
    if (i < ndst) dst[i] = (i < nsrc) ? f2bf_bits(src[i]) : (unsigned short)0;
}

// f32 copy with zero padding (for classifier bias)
__global__ void cvt_f32_pad(const float* __restrict__ src,
                            float* __restrict__ dst,
                            long nsrc, long ndst) {
    long i = (long)blockIdx.x * blockDim.x + threadIdx.x;
    if (i < ndst) dst[i] = (i < nsrc) ? src[i] : 0.0f;
}

// ---------------------------------------------------------------------------
// C[M,Nout] = A[M,K](bf16) * W[Npad,K](bf16)^T + bias   (f32 accumulate)
//
// Workgroup: 256 threads = 8 waves -> 128(M) x 256(N) tile.
// Waves: 2(M) x 4(N); each wave computes 64x64 via 4x4 fragments of
// v_wmma_f32_16x16x32_bf16 (16 WMMAs per K-step of 32).
//
// Data movement: double-buffered LDS panels filled with
// GLOBAL_LOAD_ASYNC_TO_LDS_B128 (ASYNCcnt), stored pre-swizzled into WMMA
// fragment order so every fragment read is a conflict-free ds_load_b128 at
// (base + lane*16) / (base + 512 + lane*16).
//
// LDS chunk swizzle (16-row group g, row r in group, 16B chunk j of the 64B
// K-slice):
//   A: off = g*1024 + (j>>1)*512 + ((j&1)*16 + r)*16   (frag lane = hi*16+r,
//      reads j=hi at +0 and j=2+hi at +512)
//   B: off = g*1024 + (j>>1)*512 + ((j&1)*16 + r)*16   (frag lane reads
//      j=hi at +0 and j=2+hi at +512)
// ---------------------------------------------------------------------------
#define GEMM_BM   128
#define GEMM_BN   256
#define GEMM_KS   32
#define LDS_A_BYTES 8192     // 128 rows * 64B
#define LDS_B_BYTES 16384    // 256 rows * 64B
#define LDS_BUF_BYTES (LDS_A_BYTES + LDS_B_BYTES)  // 24576
#define ASYNC_PER_WAVE 6     // 2 A-chunk instrs + 4 B-chunk instrs

__global__ __launch_bounds__(256)
void gemm_bf16_wmma(const bf16_t* __restrict__ A,
                    const bf16_t* __restrict__ W,
                    const float* __restrict__ bias,
                    float* __restrict__ C,
                    int Nout, int K) {
    __shared__ __align__(16) unsigned char smem[2 * LDS_BUF_BYTES];

    const int tid  = threadIdx.x;
    const int lane = tid & 31;
    const int r    = lane & 15;
    const int hi   = lane >> 4;
    const int wave = tid >> 5;
    const int wm   = wave & 1;   // 0..1 : 64-row slab
    const int wn   = wave >> 1;  // 0..3 : 64-col slab

    const long m_blk = (long)blockIdx.y * GEMM_BM;
    const long n_blk = (long)blockIdx.x * GEMM_BN;

    const unsigned ldsBase = (unsigned)(size_t)(&smem[0]);

    // -------- async-copy chunk assignment (16B chunks, coalesced) ----------
    // A panel: 512 chunks; thread t handles c = t + i*256, i = 0..1
    // B panel: 1024 chunks; thread t handles c = t + i*256, i = 0..3
    const bf16_t* aG[2];
    unsigned      aL[2];
#pragma unroll
    for (int i = 0; i < 2; ++i) {
        const int c   = tid + i * 256;
        const int row = c >> 2;
        const int j   = c & 3;
        aG[i] = A + (m_blk + row) * (long)K + j * 8;
        aL[i] = ldsBase + (unsigned)((row >> 4) * 1024 + (j >> 1) * 512 +
                                     ((j & 1) * 16 + (row & 15)) * 16);
    }
    const bf16_t* bG[4];
    unsigned      bL[4];
#pragma unroll
    for (int i = 0; i < 4; ++i) {
        const int c   = tid + i * 256;
        const int row = c >> 2;
        const int j   = c & 3;
        bG[i] = W + (n_blk + row) * (long)K + j * 8;
        bL[i] = ldsBase + (unsigned)(LDS_A_BYTES + (row >> 4) * 1024 +
                                     (j >> 1) * 512 +
                                     ((j & 1) * 16 + (row & 15)) * 16);
    }

    auto issue = [&](int sel, int kq) {
        const unsigned bo = (unsigned)sel * LDS_BUF_BYTES;
#pragma unroll
        for (int i = 0; i < 2; ++i)
            asm volatile("global_load_async_to_lds_b128 %0, %1, off"
                         :: "v"(aL[i] + bo), "v"(aG[i] + kq) : "memory");
#pragma unroll
        for (int i = 0; i < 4; ++i)
            asm volatile("global_load_async_to_lds_b128 %0, %1, off"
                         :: "v"(bL[i] + bo), "v"(bG[i] + kq) : "memory");
    };

    // -------- accumulators -------------------------------------------------
    v8f acc[4][4];
    const v8f vzero = {0.f, 0.f, 0.f, 0.f, 0.f, 0.f, 0.f, 0.f};
#pragma unroll
    for (int mf = 0; mf < 4; ++mf)
#pragma unroll
        for (int nf = 0; nf < 4; ++nf) acc[mf][nf] = vzero;

    // -------- software pipeline -------------------------------------------
    issue(0, 0);
    int sel = 0;
    for (int k0 = 0; k0 < K; k0 += GEMM_KS) {
        int kn = k0 + GEMM_KS;
        if (kn >= K) kn = 0;              // tail: dead prefetch, keeps wait const
        issue(sel ^ 1, kn);

        // In-order ASYNCcnt completion: <=6 outstanding => current buffer done.
        asm volatile("s_wait_asynccnt 0x6" ::: "memory");
        __syncthreads();

        const unsigned char* aBase = &smem[sel * LDS_BUF_BYTES + wm * 4096];
        const unsigned char* bBase =
            &smem[sel * LDS_BUF_BYTES + LDS_A_BYTES + wn * 4096];

        v16bf a[4], b[4];
#pragma unroll
        for (int mf = 0; mf < 4; ++mf) {
            const unsigned char* p = aBase + mf * 1024 + lane * 16;
            union { v16bf v; v8bf h[2]; } u;
            u.h[0] = *(const v8bf*)(p);
            u.h[1] = *(const v8bf*)(p + 512);
            a[mf] = u.v;
        }
#pragma unroll
        for (int nf = 0; nf < 4; ++nf) {
            const unsigned char* p = bBase + nf * 1024 + lane * 16;
            union { v16bf v; v8bf h[2]; } u;
            u.h[0] = *(const v8bf*)(p);
            u.h[1] = *(const v8bf*)(p + 512);
            b[nf] = u.v;
        }
#pragma unroll
        for (int mf = 0; mf < 4; ++mf)
#pragma unroll
            for (int nf = 0; nf < 4; ++nf)
                acc[mf][nf] = __builtin_amdgcn_wmma_f32_16x16x32_bf16(
                    false, a[mf], false, b[nf], (short)0, acc[mf][nf],
                    false, false);

        __syncthreads();   // protect next iteration's overwrite of buf sel^1
        sel ^= 1;
    }

    // -------- epilogue: bias add + store ----------------------------------
#pragma unroll
    for (int nf = 0; nf < 4; ++nf) {
        const long n = n_blk + wn * 64 + nf * 16 + r;
        if (n < Nout) {
            const float bv = bias[n];
#pragma unroll
            for (int mf = 0; mf < 4; ++mf) {
                const long m = m_blk + wm * 64 + mf * 16 + hi * 8;
#pragma unroll
                for (int e = 0; e < 8; ++e)
                    C[(m + e) * (long)Nout + n] = acc[mf][nf][e] + bv;
            }
        }
    }
}

// ---------------------------------------------------------------------------
// Per-row exact top-K mask + ReLU + bf16 convert (radix select, LDS histogram)
// ---------------------------------------------------------------------------
__global__ __launch_bounds__(256)
void topk_relu_to_bf16(const float* __restrict__ Hin,
                       unsigned short* __restrict__ Hout,
                       int Hdim, int Kkeep) {
    const int  tid = threadIdx.x;
    const long row = blockIdx.x;
    const float* hr = Hin + row * (long)Hdim;
    const int J = Hdim >> 8;  // elements per thread (H=2048 -> 8)

    float    vals[8];
    unsigned ub[8];
    for (int j = 0; j < J; ++j) {
        float v = hr[tid + (j << 8)];
        vals[j] = v;
        ub[j]   = __float_as_uint(v);
    }

    __shared__ unsigned hist[256];
    __shared__ unsigned s_prefix, s_krem, s_done;
    if (tid == 0) { s_prefix = 0u; s_krem = (unsigned)Kkeep; s_done = 0u; }
    __syncthreads();

    for (int pass = 0; pass < 4; ++pass) {
        const unsigned pfx  = s_prefix;
        const unsigned krem = s_krem;
        const unsigned done = s_done;
        hist[tid] = 0u;
        __syncthreads();

        const int shift = 24 - pass * 8;
        if (!done) {
            const unsigned hmask =
                (pass == 0) ? 0u : (0xFFFFFFFFu << (shift + 8));
            for (int j = 0; j < J; ++j)
                if (vals[j] > 0.f && (ub[j] & hmask) == pfx)
                    atomicAdd(&hist[(ub[j] >> shift) & 255u], 1u);
        }
        __syncthreads();

        if (tid == 0 && !done) {
            unsigned cum = 0;
            int sel = -1;
            for (int bin = 255; bin >= 0; --bin) {
                cum += hist[bin];
                if (cum >= krem) {
                    sel = bin;
                    s_krem   = krem - (cum - hist[bin]);
                    s_prefix = pfx | ((unsigned)bin << shift);
                    break;
                }
            }
            if (sel < 0) { s_done = 1u; s_prefix = 0u; }  // < K positives
        }
        __syncthreads();
    }

    const unsigned thr = s_prefix;
    for (int j = 0; j < J; ++j) {
        const float v = vals[j];
        const bool keep = (v > 0.f) && (ub[j] >= thr);
        Hout[row * (long)Hdim + tid + (j << 8)] =
            keep ? f2bf_bits(v) : (unsigned short)0;
    }
}

// ---------------------------------------------------------------------------
// Host-side orchestration
// ---------------------------------------------------------------------------
extern "C" void kernel_launch(void* const* d_in, const int* in_sizes, int n_in,
                              void* d_out, int out_size, void* d_ws,
                              size_t ws_size, hipStream_t stream) {
    (void)in_sizes; (void)n_in; (void)out_size; (void)ws_size;
    constexpr long B = 16384, IN = 1024, H = 2048, KK = 256;
    constexpr long NC = 1000, NCP = 1024;

    const float* x  = (const float*)d_in[0];
    const float* W0 = (const float*)d_in[1];
    const float* b0 = (const float*)d_in[2];
    const float* W1 = (const float*)d_in[3];
    const float* b1 = (const float*)d_in[4];
    const float* Wc = (const float*)d_in[5];
    const float* bc = (const float*)d_in[6];
    float* out = (float*)d_out;

    char* ws = (char*)d_ws;
    size_t off = 0;
    auto take = [&](size_t bytes) -> char* {
        char* p = ws + off;
        off = (off + bytes + 255) & ~(size_t)255;
        return p;
    };
    unsigned short* xb   = (unsigned short*)take(B * IN * 2);
    unsigned short* w0b  = (unsigned short*)take(H * IN * 2);
    unsigned short* w1b  = (unsigned short*)take(H * H * 2);
    unsigned short* wcb  = (unsigned short*)take(NCP * H * 2);
    float*          bcp  = (float*)take(NCP * 4);
    float*          htmp = (float*)take(B * H * 4);
    unsigned short* h0b  = (unsigned short*)take(B * H * 2);
    unsigned short* h1b  = (unsigned short*)take(B * H * 2);

    auto nblk = [](long n) { return (unsigned)((n + 255) / 256); };

    // Downconvert inputs to bf16 (Wc padded to 1024 rows, bc padded to 1024).
    cvt_f32_to_bf16_pad<<<nblk(B * IN), 256, 0, stream>>>(x, xb, B * IN, B * IN);
    cvt_f32_to_bf16_pad<<<nblk(H * IN), 256, 0, stream>>>(W0, w0b, H * IN, H * IN);
    cvt_f32_to_bf16_pad<<<nblk(H * H), 256, 0, stream>>>(W1, w1b, H * H, H * H);
    cvt_f32_to_bf16_pad<<<nblk(NCP * H), 256, 0, stream>>>(Wc, wcb, NC * H, NCP * H);
    cvt_f32_pad<<<nblk(NCP), 256, 0, stream>>>(bc, bcp, NC, NCP);

    // Layer 0: h = x @ W0^T + b0
    gemm_bf16_wmma<<<dim3(H / GEMM_BN, B / GEMM_BM), 256, 0, stream>>>(
        (const bf16_t*)xb, (const bf16_t*)w0b, b0, htmp, (int)H, (int)IN);
    topk_relu_to_bf16<<<(unsigned)B, 256, 0, stream>>>(htmp, h0b, (int)H, (int)KK);

    // Layer 1: h = h0 @ W1^T + b1
    gemm_bf16_wmma<<<dim3(H / GEMM_BN, B / GEMM_BM), 256, 0, stream>>>(
        (const bf16_t*)h0b, (const bf16_t*)w1b, b1, htmp, (int)H, (int)H);
    topk_relu_to_bf16<<<(unsigned)B, 256, 0, stream>>>(htmp, h1b, (int)H, (int)KK);

    // Classifier: out = h1 @ Wc^T + bc  (N=1000, weights padded to 1024 rows)
    gemm_bf16_wmma<<<dim3(NCP / GEMM_BN, B / GEMM_BM), 256, 0, stream>>>(
        (const bf16_t*)h1b, (const bf16_t*)wcb, bcp, out, (int)NC, (int)H);
}
